// AttentionLayer_14757507629111
// MI455X (gfx1250) — compile-verified, matrix operand order checked by reading, not run on hardware
//
#include <hip/hip_runtime.h>

// ---------------------------------------------------------------------------
// Fused MHA for MI455X (gfx1250, wave32, WMMA).
//   B=4, T=S=2048, D=512, H=8, HD=64
// Pipeline:
//   gemm_wt<0>(query,Wq) -> q_ws  bf16 [bh][t][64]
//   gemm_wt<0>(key,  Wk) -> k_ws  bf16 [bh][s][64]
//   gemm_wt<2>(value,Wv) -> v_ws  bf16 [bh][64][s]   (transposed for P@V)
//   flash_attn(q,k,vT)   -> ctx   bf16 [b][t][512]   (heads merged)
//   gemm_wt<3>(ctx, Wo)  -> out   f32  [b][t][512]
// Matmuls: v_wmma_f32_16x16x32_bf16, f32 accumulation.
// GEMM: block tile 128x64, wave tile 64x16 (4 acc), B-frag reused 4x.
// Flash: double-buffered GLOBAL_LOAD_ASYNC_TO_LDS_B128 staging (ASYNCcnt),
//        softmax row-reductions via DPP8 + DPP16 row_ror:8, exp2 domain.
// ---------------------------------------------------------------------------

#define DMODEL 512
#define NH     8
#define HD     64
#define SEQ    2048
#define BATCH  4
#define ROWS   (BATCH * SEQ)   // 8192

#ifndef __has_builtin
#define __has_builtin(x) 0
#endif
#if __has_builtin(__builtin_amdgcn_global_load_async_to_lds_b128) && \
    __has_builtin(__builtin_amdgcn_s_wait_asynccnt)
#define HAVE_ASYNC_LDS 1
#else
#define HAVE_ASYNC_LDS 0
#endif

#define AS1 __attribute__((address_space(1)))
#define AS3 __attribute__((address_space(3)))

typedef __attribute__((ext_vector_type(4)))  int            v4i;
typedef __attribute__((ext_vector_type(4)))  float          v4f;
typedef __attribute__((ext_vector_type(8)))  float          v8f;
typedef __attribute__((ext_vector_type(4)))  unsigned short v4u;
typedef __attribute__((ext_vector_type(8)))  unsigned short v8u;
typedef __attribute__((ext_vector_type(16))) unsigned short v16u;
typedef __attribute__((ext_vector_type(16))) __bf16         v16bf;

__device__ __forceinline__ unsigned short f2bf(float f) {
  unsigned int u = __float_as_uint(f);
  u += 0x7FFFu + ((u >> 16) & 1u);            // round-to-nearest-even
  return (unsigned short)(u >> 16);
}

__device__ __forceinline__ v8f wmma_bf16(v16u a, v16u b, v8f c) {
  return __builtin_amdgcn_wmma_f32_16x16x32_bf16(
      /*neg_a=*/false, __builtin_bit_cast(v16bf, a),
      /*neg_b=*/false, __builtin_bit_cast(v16bf, b),
      /*c_mod=*/(short)0, c, /*reuse_a=*/false, /*reuse_b=*/false);
}

union FragU { v16u v; struct { v8u lo, hi; } p; };

// A/B 16-bit fragment = two contiguous 16B chunks per lane:
//   elements 0-7 at k0, elements 8-15 at k1 (k0 = kbase + half*8, k1 = k0+16)
__device__ __forceinline__ v16u frag_ld(const unsigned short* row, int k0, int k1) {
  FragU f;
  f.p.lo = *(const v8u*)(row + k0);
  f.p.hi = *(const v8u*)(row + k1);
  return f.v;
}

// ---- cross-lane row reductions (16-lane rows) via DPP, no LDS traffic -----
#define DPP8_SEL(a,b,c,d,e,f,g,h) \
  ((a)|((b)<<3)|((c)<<6)|((d)<<9)|((e)<<12)|((f)<<15)|((g)<<18)|((h)<<21))

template <int SEL>
__device__ __forceinline__ float dpp8_f(float x) {
  return __int_as_float(__builtin_amdgcn_mov_dpp8(__float_as_int(x), SEL));
}
__device__ __forceinline__ float row_xor8_f(float x) {  // DPP16 row_ror:8 == xor 8
  int i = __float_as_int(x);
  return __int_as_float(__builtin_amdgcn_update_dpp(i, i, 0x128, 0xF, 0xF, false));
}
__device__ __forceinline__ float rowmax16(float x) {
  x = fmaxf(x, dpp8_f<DPP8_SEL(1,0,3,2,5,4,7,6)>(x));   // xor 1
  x = fmaxf(x, dpp8_f<DPP8_SEL(2,3,0,1,6,7,4,5)>(x));   // xor 2
  x = fmaxf(x, dpp8_f<DPP8_SEL(4,5,6,7,0,1,2,3)>(x));   // xor 4
  x = fmaxf(x, row_xor8_f(x));                           // xor 8
  return x;
}
__device__ __forceinline__ float rowsum16(float x) {
  x += dpp8_f<DPP8_SEL(1,0,3,2,5,4,7,6)>(x);
  x += dpp8_f<DPP8_SEL(2,3,0,1,6,7,4,5)>(x);
  x += dpp8_f<DPP8_SEL(4,5,6,7,0,1,2,3)>(x);
  x += row_xor8_f(x);
  return x;
}

// ---------------------------------------------------------------------------
// Tiled GEMM:  Y[i][j] = sum_k X[i][k] * W[j][k] + bias[j]
// Block tile 128x64, K-step 64, 8 waves (2 along M x 4 along N).
// Wave tile 64x16: 4 accumulators; per k-step 1 B-frag + 4 A-frags -> 4 WMMA.
// MODE 0: bf16 out, [bh][t][hd]   (Q / K per-head layout)
// MODE 2: bf16 out, [bh][hd][s]   (V transposed per-head layout)
// MODE 3: f32 out, row-major [ROWS][D]  (final projection -> d_out)
// ---------------------------------------------------------------------------
template <int MODE, typename XT>
__global__ __launch_bounds__(256)
void gemm_wt(const XT* __restrict__ X, const float* __restrict__ W,
             const float* __restrict__ bias, void* __restrict__ OUT) {
  __shared__ __align__(16) unsigned short Xs[128][72];  // +8 pad: 16B-aligned rows
  __shared__ __align__(16) unsigned short Ws[64][72];

  const int tid = threadIdx.x;
  const int lane = tid & 31, lane16 = lane & 15, half = (lane >> 4) & 1;
  const int wid = tid >> 5;
  const int m0 = blockIdx.y * 128;
  const int n0 = blockIdx.x * 64;
  const int wave_m = (wid & 1) * 64;
  const int wave_n = (wid >> 1) * 16;

  const v8f vz = {0.f, 0.f, 0.f, 0.f, 0.f, 0.f, 0.f, 0.f};
  v8f acc[4] = {vz, vz, vz, vz};

  for (int k0 = 0; k0 < DMODEL; k0 += 64) {
    // ---- stage X tile (128 rows x 64 k) as bf16 ----
    if constexpr (sizeof(XT) == 4) {
#pragma unroll
      for (int i = 0; i < 8; ++i) {
        int chunk = tid + i * 256;               // 2048 float4 chunks
        int row = chunk >> 4, k4 = (chunk & 15) * 4;
        v4f x = *(const v4f*)((const float*)X + (size_t)(m0 + row) * DMODEL + k0 + k4);
        v4u h;
        h[0] = f2bf(x[0]); h[1] = f2bf(x[1]); h[2] = f2bf(x[2]); h[3] = f2bf(x[3]);
        *(v4u*)(&Xs[row][k4]) = h;
      }
    } else {                                     // bf16 input (ctx)
#pragma unroll
      for (int i = 0; i < 4; ++i) {
        int chunk = tid + i * 256;               // 1024 v8u chunks
        int row = chunk >> 3, k8 = (chunk & 7) * 8;
        *(v8u*)(&Xs[row][k8]) =
            *(const v8u*)((const unsigned short*)X + (size_t)(m0 + row) * DMODEL + k0 + k8);
      }
    }
    // ---- stage W tile (64 out-cols x 64 k) as bf16 ----
#pragma unroll
    for (int i = 0; i < 4; ++i) {
      int chunk = tid + i * 256;                 // 1024 float4 chunks
      int row = chunk >> 4, k4 = (chunk & 15) * 4;
      v4f x = *(const v4f*)(W + (size_t)(n0 + row) * DMODEL + k0 + k4);
      v4u h;
      h[0] = f2bf(x[0]); h[1] = f2bf(x[1]); h[2] = f2bf(x[2]); h[3] = f2bf(x[3]);
      *(v4u*)(&Ws[row][k4]) = h;
    }
    __syncthreads();

#pragma unroll
    for (int kk = 0; kk < 64; kk += 32) {
      const int c0 = kk + half * 8, c1 = kk + 16 + half * 8;
      v16u b = frag_ld(&Ws[wave_n + lane16][0], c0, c1);
#pragma unroll
      for (int mt = 0; mt < 4; ++mt) {
        v16u a = frag_ld(&Xs[wave_m + mt * 16 + lane16][0], c0, c1);
        acc[mt] = wmma_bf16(a, b, acc[mt]);
      }
    }
    __syncthreads();
  }

  // ---- epilogue: bias + layout-specific store ----
  const int j = n0 + wave_n + lane16;
  const float bj = bias[j];
#pragma unroll
  for (int mt = 0; mt < 4; ++mt) {
#pragma unroll
    for (int r = 0; r < 8; ++r) {
      const int i = m0 + wave_m + mt * 16 + r + 8 * half;  // global row
      const float val = acc[mt][r] + bj;
      if constexpr (MODE == 3) {
        ((float*)OUT)[(size_t)i * DMODEL + j] = val;
      } else {
        const int bb = i >> 11, t = i & (SEQ - 1);
        const int h = j >> 6, d = j & (HD - 1);
        size_t idx;
        if constexpr (MODE == 0)
          idx = ((size_t)(bb * NH + h) * SEQ + t) * HD + d;
        else
          idx = ((size_t)(bb * NH + h) * HD + d) * SEQ + t;
        ((unsigned short*)OUT)[idx] = f2bf(val);
      }
    }
  }
}

// ---------------------------------------------------------------------------
// Flash attention: 1 block per (bh, 64 query rows), 4 waves, 16 rows/wave.
// S-tile = 64; K (64x64) and V^T (64x64) tiles double-buffered in LDS,
// staged with async global->LDS when available. Softmax in exp2 domain.
// ---------------------------------------------------------------------------
__device__ __forceinline__ void stage_tiles(const unsigned short* __restrict__ K,
                                            const unsigned short* __restrict__ VT,
                                            unsigned short (*Ksb)[72],
                                            unsigned short (*Vsb)[72],
                                            int bh, int s0, int tid) {
#pragma unroll
  for (int i = 0; i < 4; ++i) {
    int chunk = tid + i * 128;                   // 512 16B chunks per tile
    int row = chunk >> 3, k8 = (chunk & 7) * 8;  // K: row=s, k8=d ; V: row=d, k8=s
    const unsigned short* gk = K  + ((size_t)bh * SEQ + s0 + row) * HD + k8;
    const unsigned short* gv = VT + ((size_t)bh * HD + row) * SEQ + s0 + k8;
#if HAVE_ASYNC_LDS
    __builtin_amdgcn_global_load_async_to_lds_b128((AS1 v4i*)gk,
                                                   (AS3 v4i*)&Ksb[row][k8], 0, 0);
    __builtin_amdgcn_global_load_async_to_lds_b128((AS1 v4i*)gv,
                                                   (AS3 v4i*)&Vsb[row][k8], 0, 0);
#else
    *(v8u*)(&Ksb[row][k8]) = *(const v8u*)gk;
    *(v8u*)(&Vsb[row][k8]) = *(const v8u*)gv;
#endif
  }
}

__global__ __launch_bounds__(128)
void flash_attn(const unsigned short* __restrict__ Q,
                const unsigned short* __restrict__ K,
                const unsigned short* __restrict__ VT,
                unsigned short* __restrict__ CTX) {
  __shared__ __align__(16) unsigned short Ks[2][64][72];   // K tiles  [s][d]
  __shared__ __align__(16) unsigned short Vs[2][64][72];   // V^T tiles [d][s]
  __shared__ __align__(16) unsigned short Ps[4][16][72];   // per-wave P bounce

  const int tid = threadIdx.x;
  const int lane = tid & 31, lane16 = lane & 15, half = (lane >> 4) & 1;
  const int wid = tid >> 5;
  const int bh = blockIdx.y;                 // b*NH + h
  const int t0 = blockIdx.x * 64;
  const float SCL = 0.125f * 1.44269504089f; // 1/sqrt(HD) * log2(e): exp2 domain

  // Q fragments resident in VGPRs (row = lane16's query row, k = d in 0..63)
  const unsigned short* qrow = Q + ((size_t)bh * SEQ + t0 + wid * 16 + lane16) * HD;
  const v16u qa0 = frag_ld(qrow, half * 8, 16 + half * 8);
  const v16u qa1 = frag_ld(qrow, 32 + half * 8, 48 + half * 8);

  const v8f vz = {0.f, 0.f, 0.f, 0.f, 0.f, 0.f, 0.f, 0.f};
  v8f o[4] = {vz, vz, vz, vz};               // O acc: d-tiles 0..3
  float m_i[8], l_i[8];
#pragma unroll
  for (int r = 0; r < 8; ++r) { m_i[r] = -3.0e38f; l_i[r] = 0.f; }

  stage_tiles(K, VT, Ks[0], Vs[0], bh, 0, tid);
#if HAVE_ASYNC_LDS
  __builtin_amdgcn_s_wait_asynccnt(0);
#endif
  __syncthreads();

  for (int s0 = 0; s0 < SEQ; s0 += 64) {
    const int buf = (s0 >> 6) & 1;
    if (s0 + 64 < SEQ)   // prefetch next tiles into the other buffer
      stage_tiles(K, VT, Ks[buf ^ 1], Vs[buf ^ 1], bh, s0 + 64, tid);

    // ---- scores S = Q K^T : four 16x16 tiles over s, chained over d ----
    v8f c[4];
#pragma unroll
    for (int nt = 0; nt < 4; ++nt) {
      const unsigned short* krow = &Ks[buf][nt * 16 + lane16][0];
      v8f cc = vz;
      cc = wmma_bf16(qa0, frag_ld(krow, half * 8, 16 + half * 8), cc);
      cc = wmma_bf16(qa1, frag_ld(krow, 32 + half * 8, 48 + half * 8), cc);
      c[nt] = cc;
    }

    // ---- online softmax (rows r+8*half, 4 cols/lane), exp2 domain ----
#pragma unroll
    for (int r = 0; r < 8; ++r) {
      float v0 = c[0][r] * SCL, v1 = c[1][r] * SCL;
      float v2 = c[2][r] * SCL, v3 = c[3][r] * SCL;
      float tm = rowmax16(fmaxf(fmaxf(v0, v1), fmaxf(v2, v3)));
      float nm = fmaxf(m_i[r], tm);
      float p0 = exp2f(v0 - nm), p1 = exp2f(v1 - nm);
      float p2 = exp2f(v2 - nm), p3 = exp2f(v3 - nm);
      float rs = rowsum16((p0 + p1) + (p2 + p3));
      float corr = exp2f(m_i[r] - nm);
      l_i[r] = l_i[r] * corr + rs;
      m_i[r] = nm;
      o[0][r] *= corr; o[1][r] *= corr; o[2][r] *= corr; o[3][r] *= corr;
      const int prow = r + 8 * half;            // C-layout -> A-layout bounce
      Ps[wid][prow][lane16]      = f2bf(p0);
      Ps[wid][prow][16 + lane16] = f2bf(p1);
      Ps[wid][prow][32 + lane16] = f2bf(p2);
      Ps[wid][prow][48 + lane16] = f2bf(p3);
    }

    // ---- O += P @ V  (k = 64 s-values, 4 d-tiles) ----
    const unsigned short* prow = &Ps[wid][lane16][0];
    v16u pa0 = frag_ld(prow, half * 8, 16 + half * 8);
    v16u pa1 = frag_ld(prow, 32 + half * 8, 48 + half * 8);
#pragma unroll
    for (int dt = 0; dt < 4; ++dt) {
      const unsigned short* vrow = &Vs[buf][dt * 16 + lane16][0];
      o[dt] = wmma_bf16(pa0, frag_ld(vrow, half * 8, 16 + half * 8), o[dt]);
      o[dt] = wmma_bf16(pa1, frag_ld(vrow, 32 + half * 8, 48 + half * 8), o[dt]);
    }

#if HAVE_ASYNC_LDS
    __builtin_amdgcn_s_wait_asynccnt(0);
#endif
    __syncthreads();
  }

  // ---- normalize and store merged-head context (bf16) ----
  const int b_ = bh >> 3, h = bh & 7;
#pragma unroll
  for (int r = 0; r < 8; ++r) {
    const float inv = 1.0f / l_i[r];
    const int t = t0 + wid * 16 + r + 8 * half;
    const size_t base = ((size_t)b_ * SEQ + t) * DMODEL + h * HD;
#pragma unroll
    for (int dt = 0; dt < 4; ++dt)
      CTX[base + dt * 16 + lane16] = f2bf(o[dt][r] * inv);
  }
}

// ---------------------------------------------------------------------------
extern "C" void kernel_launch(void* const* d_in, const int* in_sizes, int n_in,
                              void* d_out, int out_size, void* d_ws, size_t ws_size,
                              hipStream_t stream) {
  (void)in_sizes; (void)n_in; (void)out_size; (void)ws_size;
  const float* query = (const float*)d_in[0];
  const float* key   = (const float*)d_in[1];
  const float* value = (const float*)d_in[2];
  const float* Wq = (const float*)d_in[3];  const float* bq = (const float*)d_in[4];
  const float* Wk = (const float*)d_in[5];  const float* bk = (const float*)d_in[6];
  const float* Wv = (const float*)d_in[7];  const float* bv = (const float*)d_in[8];
  const float* Wo = (const float*)d_in[9];  const float* bo = (const float*)d_in[10];

  // workspace: 4 bf16 buffers of ROWS*DMODEL (8 MiB each, 32 MiB total)
  unsigned short* q_ws = (unsigned short*)d_ws;
  unsigned short* k_ws = q_ws + (size_t)ROWS * DMODEL;
  unsigned short* v_ws = k_ws + (size_t)ROWS * DMODEL;
  unsigned short* ctx  = v_ws + (size_t)ROWS * DMODEL;

  dim3 blk(256);
  dim3 grd(DMODEL / 64, ROWS / 128);  // (8, 64)
  gemm_wt<0, float><<<grd, blk, 0, stream>>>(query, Wq, bq, q_ws);
  gemm_wt<0, float><<<grd, blk, 0, stream>>>(key,   Wk, bk, k_ws);
  gemm_wt<2, float><<<grd, blk, 0, stream>>>(value, Wv, bv, v_ws);

  flash_attn<<<dim3(SEQ / 64, BATCH * NH), dim3(128), 0, stream>>>(q_ws, k_ws, v_ws, ctx);

  gemm_wt<3, unsigned short><<<grd, blk, 0, stream>>>(ctx, Wo, bo, d_out);
}
